// Get_Correlation_30382598652330
// MI455X (gfx1250) — compile-verified, matrix operand order checked by reading, not run on hardware
//
#include <hip/hip_runtime.h>
#include <hip/hip_bf16.h>

// ---------------------------------------------------------------------------
// Problem constants (from reference): B=4, C=1024, T=16, H=W=28
// ---------------------------------------------------------------------------
#define BB   4
#define CC   1024
#define TT   16
#define HH   28
#define WW   28
#define HWP  784          // H*W
#define BT   64           // B*T
#define RC   64           // C/16
#define QP   800          // HW padded up to multiple of 32 (K of GEMM3)

typedef __attribute__((ext_vector_type(16))) __bf16 v16bf;
typedef __attribute__((ext_vector_type(8)))  float  v8f;

union Frag { v16bf v; unsigned u[8]; };

__device__ __forceinline__ unsigned short f2bf(float f) {
    unsigned u = __float_as_uint(f);
    u += 0x7fffu + ((u >> 16) & 1u);          // round-to-nearest-even
    return (unsigned short)(u >> 16);
}
__device__ __forceinline__ unsigned pack2bf(float lo, float hi) {
    return (unsigned)f2bf(lo) | ((unsigned)f2bf(hi) << 16);
}
__device__ __forceinline__ float sigm_m05(float x) {
    return 1.0f / (1.0f + __expf(-x)) - 0.5f;
}
__device__ __forceinline__ v8f vzero() {
    v8f z = {0.f, 0.f, 0.f, 0.f, 0.f, 0.f, 0.f, 0.f};
    return z;
}

// ---------------------------------------------------------------------------
// Fragment loads from per-lane pointers (addresses advanced by the caller).
// A: row-major bf16; lane pointer = base + (m0+lr)*(ld/2) + half*4;
//    dwords {0..3} and {8..11}  -> 2x b128.  Advance +16 dwords per k-step.
// B: fragment-swizzled pair-packed; lane pointer =
//    base + ((kt*2+half)*N + n0+lr)*8; dwords {0..7} -> 2x b128 (32B aligned).
//    Advance +16*N dwords per k-step.
// ---------------------------------------------------------------------------
__device__ __forceinline__ v16bf frag_from_a(const unsigned* pA) {
    Frag f;
#pragma unroll
    for (int j = 0; j < 4; ++j) { f.u[j] = pA[j]; f.u[j + 4] = pA[8 + j]; }
    return f.v;
}
__device__ __forceinline__ v16bf frag_from_b(const unsigned* pB) {
    Frag f;
#pragma unroll
    for (int j = 0; j < 8; ++j) f.u[j] = pB[j];
    return f.v;
}

__device__ __forceinline__ v8f wmma_bf16(v16bf a, v16bf b, v8f c) {
    return __builtin_amdgcn_wmma_f32_16x16x32_bf16(false, a, false, b,
                                                   (short)0, c, false, false);
}

// ---------------------------------------------------------------------------
// K0: cast w_down2 (C*C) and w_down (RC*C) to bf16 row-major
// ---------------------------------------------------------------------------
__global__ __launch_bounds__(256) void k_cast_w(const float* w2, const float* wd,
                                                unsigned short* w2b, unsigned short* wdb) {
    int i = blockIdx.x * 256 + threadIdx.x;
    if (i < CC * CC) w2b[i] = f2bf(w2[i]);
    if (i < RC * CC) wdb[i] = f2bf(wd[i]);
}

// ---------------------------------------------------------------------------
// K1: tiled transpose of x -> xbT [bt][p][c] (bf16, A-side)
//                         and xbp (fragment-swizzled pair-packed, B-side)
// ---------------------------------------------------------------------------
__global__ __launch_bounds__(256) void k_transpose(const float* __restrict__ x,
                                                   unsigned short* __restrict__ xbT,
                                                   unsigned* __restrict__ xbp) {
    const int bt = blockIdx.x, b = bt / TT, t = bt % TT;
    const int c0 = blockIdx.y * 32, p0 = blockIdx.z * 32;   // c0: one 32-wide K tile
    __shared__ float tile[32][33];
    const int tx = threadIdx.x, ty = threadIdx.y;           // (32, 8)
#pragma unroll
    for (int i = 0; i < 4; ++i) {
        int c = c0 + ty + i * 8, p = p0 + tx;
        tile[ty + i * 8][tx] = (p < HWP)
            ? x[((size_t)(b * CC + c) * TT + t) * HWP + p] : 0.0f;
    }
    __syncthreads();
#pragma unroll
    for (int i = 0; i < 4; ++i) {                           // xbT: contiguous over c
        int pr = ty + i * 8, p = p0 + pr;
        if (p < HWP)
            xbT[((size_t)bt * HWP + p) * CC + c0 + tx] = f2bf(tile[tx][pr]);
    }
    // xbp fragment-swizzled: thread -> (p_local, j); writes coalesced dwords
    const int idx = ty * 32 + tx;
    const int pj = idx >> 3, j = idx & 7;
    const int p = p0 + pj;
    if (p < HWP) {
        const int kt = c0 >> 5;
#pragma unroll
        for (int half = 0; half < 2; ++half) {
            int k2l = (j & 3) + ((j & 4) << 1) + half * 4;  // local K-pair 0..15
            unsigned d = pack2bf(tile[2 * k2l][pj], tile[2 * k2l + 1][pj]);
            xbp[((((size_t)bt * 32 + kt) * 2 + half) * HWP + p) * 8 + j] = d;
        }
    }
}

// ---------------------------------------------------------------------------
// K2: zero the K-padding columns (784..799) of x2row
// ---------------------------------------------------------------------------
__global__ __launch_bounds__(256) void k_pad_x2row(unsigned* x2row_dw) {
    int idx = blockIdx.x * 256 + threadIdx.x;               // BT*C*8
    if (idx >= BT * CC * 8) return;
    int j = idx & 7, row = idx >> 3;
    x2row_dw[(size_t)row * (QP / 2) + (HWP / 2) + j] = 0u;
}

// ---------------------------------------------------------------------------
// Store one 16x16 f32 D tile of x2: row-major bf16 (ld=QP) + swizzled pack.
// D layout: value r of lane -> (m = r + 8*half, n = lane&15).
// ---------------------------------------------------------------------------
__device__ __forceinline__ void store_x2_tile(const v8f& acc, int bt, int mt, int n, int half,
                                              unsigned short* __restrict__ x2row,
                                              unsigned* __restrict__ x2bp) {
    const int m0 = mt * 16;
#pragma unroll
    for (int r = 0; r < 8; ++r) {
        int c = m0 + half * 8 + r;
        x2row[((size_t)bt * CC + c) * QP + n] = f2bf(acc[r]);
    }
    // c2 = mt*8 + half*4 + jj  ->  kt = mt>>1, j = jj | ((mt&1)<<2), half_dst = half
    size_t basei = ((((size_t)bt * 32 + (mt >> 1)) * 2 + half) * HWP + n) * 8
                   + ((mt & 1) << 2);
#pragma unroll
    for (int jj = 0; jj < 4; ++jj)
        x2bp[basei + jj] = pack2bf(acc[2 * jj], acc[2 * jj + 1]);
}

// ---------------------------------------------------------------------------
// K3: x2 = w_down2 * x  (per bt: M=1024, N=784, K=1024), 2-wide M blocking,
//     1-stage software pipeline, pointer-increment addressing.
// ---------------------------------------------------------------------------
__global__ __launch_bounds__(256) void k_gemm_x2(const unsigned* __restrict__ w2b_dw,
                                                 const unsigned* __restrict__ xbp,
                                                 unsigned short* __restrict__ x2row,
                                                 unsigned* __restrict__ x2bp) {
    const int bt = blockIdx.x, chunk = blockIdx.y;          // chunk: 0..6
    const int wave = threadIdx.x >> 5, lane = threadIdx.x & 31;
    const int half = lane >> 4, lr = lane & 15;
    const unsigned* Bbase = xbp + (size_t)bt * (CC / 2) * HWP;
    for (int task = wave; task < 32 * 7; task += 8) {
        const int mt2 = task / 7, nt = chunk * 7 + task % 7;
        const int m0 = mt2 * 32, n0 = nt * 16;
        const unsigned* pA0 = w2b_dw + (size_t)(m0 + lr) * (CC / 2) + half * 4;
        const unsigned* pA1 = pA0 + (size_t)16 * (CC / 2);
        const unsigned* pB  = Bbase + ((size_t)half * HWP + n0 + lr) * 8;
        v8f acc0 = vzero(), acc1 = vzero();
        v16bf b_c  = frag_from_b(pB);
        v16bf a0_c = frag_from_a(pA0);
        v16bf a1_c = frag_from_a(pA1);
        for (int ks = 0; ks < 31; ++ks) {
            pB += 16 * HWP; pA0 += 16; pA1 += 16;
            v16bf b_n  = frag_from_b(pB);
            v16bf a0_n = frag_from_a(pA0);
            v16bf a1_n = frag_from_a(pA1);
            acc0 = wmma_bf16(a0_c, b_c, acc0);
            acc1 = wmma_bf16(a1_c, b_c, acc1);
            b_c = b_n; a0_c = a0_n; a1_c = a1_n;
        }
        acc0 = wmma_bf16(a0_c, b_c, acc0);
        acc1 = wmma_bf16(a1_c, b_c, acc1);
        const int n = n0 + lr;
        store_x2_tile(acc0, bt, mt2 * 2,     n, half, x2row, x2bp);
        store_x2_tile(acc1, bt, mt2 * 2 + 1, n, half, x2row, x2bp);
    }
}

// ---------------------------------------------------------------------------
// K4: xd = w_down * x  (per bt: M=64, N=784, K=1024) -> f32 [b][r][t][p]
// ---------------------------------------------------------------------------
__global__ __launch_bounds__(256) void k_gemm_xd(const unsigned* __restrict__ wdb_dw,
                                                 const unsigned* __restrict__ xbp,
                                                 float* __restrict__ xd) {
    const int bt = blockIdx.x, b = bt / TT, t = bt % TT;
    const int wave = threadIdx.x >> 5, lane = threadIdx.x & 31;
    const int half = lane >> 4, lr = lane & 15;
    const unsigned* Bbase = xbp + (size_t)bt * (CC / 2) * HWP;
    for (int task = wave; task < 4 * 49; task += 8) {
        const int mt = task / 49, nt = task % 49;
        const unsigned* pA = wdb_dw + (size_t)(mt * 16 + lr) * (CC / 2) + half * 4;
        const unsigned* pB = Bbase + ((size_t)half * HWP + nt * 16 + lr) * 8;
        v8f acc = vzero();
        v16bf b_c = frag_from_b(pB);
        v16bf a_c = frag_from_a(pA);
        for (int ks = 0; ks < 31; ++ks) {
            pB += 16 * HWP; pA += 16;
            v16bf b_n = frag_from_b(pB);
            v16bf a_n = frag_from_a(pA);
            acc = wmma_bf16(a_c, b_c, acc);
            b_c = b_n; a_c = a_n;
        }
        acc = wmma_bf16(a_c, b_c, acc);
        const int n = nt * 16 + lr;
#pragma unroll
        for (int r = 0; r < 8; ++r) {
            int rch = mt * 16 + half * 8 + r;
            xd[((size_t)(b * RC + rch) * TT + t) * HWP + n] = acc[r];
        }
    }
}

// ---------------------------------------------------------------------------
// K5: agg = sum_d weights[d] * (dwconv3d_d(xd) + b_d)   (VALU; <1% of FLOPs)
// ---------------------------------------------------------------------------
__global__ __launch_bounds__(256) void k_conv(const float* __restrict__ xd,
                                              const float* w1, const float* b1,
                                              const float* w2, const float* b2,
                                              const float* w3, const float* b3,
                                              const float* wts,
                                              float* __restrict__ agg) {
    int idx = blockIdx.x * 256 + threadIdx.x;               // B*RC*T*HW
    if (idx >= BB * RC * TT * HWP) return;
    const int p = idx % HWP, t = (idx / HWP) % TT, r = (idx / (HWP * TT)) % RC,
              b = idx / (HWP * TT * RC);
    const int h = p / WW, w = p % WW;
    const float* wk0 = w1 + r * 81;
    const float* wk1 = w2 + r * 81;
    const float* wk2 = w3 + r * 81;
    float s0 = 0.f, s1 = 0.f, s2 = 0.f;
    for (int kt = 0; kt < 9; ++kt) {
        int tt = t + kt - 4;
        if (tt < 0 || tt >= TT) continue;
        const float* xs = xd + ((size_t)(b * RC + r) * TT + tt) * HWP;
        for (int kh = 0; kh < 3; ++kh)
            for (int kw = 0; kw < 3; ++kw) {
                int tap = (kt * 3 + kh) * 3 + kw;
                int h1 = h + (kh - 1), w1i = w + (kw - 1);
                int h2 = h + (kh - 1) * 2, w2i = w + (kw - 1) * 2;
                int h3 = h + (kh - 1) * 3, w3i = w + (kw - 1) * 3;
                if (h1 >= 0 && h1 < HH && w1i >= 0 && w1i < WW) s0 += wk0[tap] * xs[h1 * WW + w1i];
                if (h2 >= 0 && h2 < HH && w2i >= 0 && w2i < WW) s1 += wk1[tap] * xs[h2 * WW + w2i];
                if (h3 >= 0 && h3 < HH && w3i >= 0 && w3i < WW) s2 += wk2[tap] * xs[h3 * WW + w3i];
            }
    }
    agg[idx] = wts[0] * (s0 + b1[r]) + wts[1] * (s1 + b2[r]) + wts[2] * (s2 + b3[r]);
}

// ---------------------------------------------------------------------------
// K6: fused affinity + sigmoid + feature GEMM per (bt, 16-wide P tile).
//   GEMM2: aff(16x784) = X^T(16xC) * X2s(CxQ), both temporal shifts;
//          batch-loaded B fragments (7 per k-step), clamped tail (uniform).
//   G = (sigmoid(aff)-0.5)*weights2[dir] -> fragment-swizzled bf16 in LDS.
//   GEMM3: feat(Cx16) += X2s(CxQP) * G^T(QPx16), 2-wide M, pipelined.
// ---------------------------------------------------------------------------
__global__ __launch_bounds__(256) void k_affinity(const unsigned* __restrict__ xbT_dw,
                                                  const unsigned* __restrict__ x2bp,
                                                  const unsigned* __restrict__ x2row_dw,
                                                  const float* __restrict__ weights2,
                                                  float* __restrict__ out) {
    const int bt = blockIdx.x, b = bt / TT, t = bt % TT;
    const int p0 = blockIdx.y * 16;
    __shared__ __align__(32) unsigned sG[2][QP / 32][2][16][8];   // 51.2 KB
    const int tid = threadIdx.x;
    for (int i = tid; i < 2 * (QP / 32) * 2 * 16 * 8; i += 256) (&sG[0][0][0][0][0])[i] = 0u;
    __syncthreads();

    const int wave = tid >> 5, lane = tid & 31, half = lane >> 4, lr = lane & 15;
    const int tn = (t + 1 < TT) ? t + 1 : TT - 1;
    const int tp = (t > 0) ? t - 1 : 0;
    const int btdir[2] = { b * TT + tn, b * TT + tp };
    const unsigned* Abase = xbT_dw + (size_t)bt * HWP * (CC / 2);

    for (int dir = 0; dir < 2; ++dir) {
        const unsigned* Bbase = x2bp + (size_t)btdir[dir] * (CC / 2) * HWP;
        const float w2d = weights2[dir];
        const unsigned* pA = Abase + (size_t)(p0 + lr) * (CC / 2) + half * 4;
        const unsigned* pB[7];
#pragma unroll
        for (int i = 0; i < 7; ++i) {
            int qt = wave + 8 * i;
            if (qt > 48) qt = 48;                            // tail waves redo tile 48
            pB[i] = Bbase + ((size_t)half * HWP + qt * 16 + lr) * 8;
        }
        v8f acc[7];
#pragma unroll
        for (int i = 0; i < 7; ++i) acc[i] = vzero();
        for (int ks = 0; ks < 32; ++ks) {
            __builtin_prefetch(pA + 16, 0, 1);
            v16bf a = frag_from_a(pA); pA += 16;
            v16bf bf[7];
#pragma unroll
            for (int i = 0; i < 7; ++i) { bf[i] = frag_from_b(pB[i]); pB[i] += 16 * HWP; }
#pragma unroll
            for (int i = 0; i < 7; ++i) acc[i] = wmma_bf16(a, bf[i], acc[i]);
        }
        // sigmoid-0.5, scale, cross-lane pack q-pairs -> swizzled LDS
#pragma unroll
        for (int i = 0; i < 7; ++i) {
            int qt = wave + 8 * i;
            if (qt >= 49) break;                             // discard duplicate tile
#pragma unroll
            for (int r = 0; r < 8; ++r) {
                float g = sigm_m05(acc[i][r]) * w2d;
                float gn = __shfl_xor(g, 1, 32);
                if (!(lane & 1)) {
                    int q2 = qt * 8 + (lr >> 1);             // global K-pair index
                    int ktl = q2 >> 4, k2l = q2 & 15;
                    int hd = (k2l >> 2) & 1;
                    int jd = (k2l & 3) | ((k2l & 8) >> 1);
                    sG[dir][ktl][hd][r + 8 * half][jd] =
                        (unsigned)f2bf(g) | ((unsigned)f2bf(gn) << 16);
                }
            }
        }
    }
    __syncthreads();

    for (int mt2 = wave; mt2 < 32; mt2 += 8) {
        const int m0 = mt2 * 32;
        v8f acc0 = vzero(), acc1 = vzero();
        for (int dir = 0; dir < 2; ++dir) {
            const unsigned* A2 = x2row_dw + (size_t)btdir[dir] * CC * (QP / 2);
            const unsigned* pA0 = A2 + (size_t)(m0 + lr) * (QP / 2) + half * 4;
            const unsigned* pA1 = pA0 + (size_t)16 * (QP / 2);
            const unsigned* pL  = &sG[dir][0][half][lr][0];
            v16bf b_c  = frag_from_b(pL);
            v16bf a0_c = frag_from_a(pA0);
            v16bf a1_c = frag_from_a(pA1);
            for (int ks = 0; ks < 24; ++ks) {
                pL += 2 * 16 * 8; pA0 += 16; pA1 += 16;
                v16bf b_n  = frag_from_b(pL);
                v16bf a0_n = frag_from_a(pA0);
                v16bf a1_n = frag_from_a(pA1);
                acc0 = wmma_bf16(a0_c, b_c, acc0);
                acc1 = wmma_bf16(a1_c, b_c, acc1);
                b_c = b_n; a0_c = a0_n; a1_c = a1_n;
            }
            acc0 = wmma_bf16(a0_c, b_c, acc0);
            acc1 = wmma_bf16(a1_c, b_c, acc1);
        }
        const int p = p0 + lr;
#pragma unroll
        for (int r = 0; r < 8; ++r) {
            int c0 = m0 + half * 8 + r;
            out[((size_t)(b * CC + c0) * TT + t) * HWP + p] = acc0[r];
            int c1 = m0 + 16 + half * 8 + r;
            out[((size_t)(b * CC + c1) * TT + t) * HWP + p] = acc1[r];
        }
    }
}

// ---------------------------------------------------------------------------
// K7: out *= sigmoid(w_back * agg) - 0.5   (rc=64 dot per element, VALU)
// ---------------------------------------------------------------------------
__global__ __launch_bounds__(256) void k_final(float* __restrict__ out,
                                               const float* __restrict__ agg,
                                               const float* __restrict__ wback) {
    size_t idx = (size_t)blockIdx.x * 256 + threadIdx.x;
    if (idx >= (size_t)BB * CC * TT * HWP) return;
    const int p = (int)(idx % HWP), t = (int)((idx / HWP) % TT),
              c = (int)((idx / (HWP * TT)) % CC), b = (int)(idx / ((size_t)HWP * TT * CC));
    const float* arow = agg + ((size_t)b * RC * TT + t) * HWP + p;
    const float* wrow = wback + (size_t)c * RC;
    float g = 0.f;
#pragma unroll 8
    for (int r = 0; r < RC; ++r) g += wrow[r] * arow[(size_t)r * TT * HWP];
    out[idx] *= sigm_m05(g);
}

// ---------------------------------------------------------------------------
// Workspace layout (bytes, all 256-aligned)
// ---------------------------------------------------------------------------
#define OFF_XBT   ((size_t)0)
#define SZ_XBT    ((size_t)BT * HWP * CC * 2)        // 102,760,448
#define OFF_XBP   (OFF_XBT + SZ_XBT)
#define SZ_XBP    ((size_t)BT * (CC / 2) * HWP * 4)  // 102,760,448
#define OFF_X2BP  (OFF_XBP + SZ_XBP)
#define SZ_X2BP   SZ_XBP
#define OFF_X2ROW (OFF_X2BP + SZ_X2BP)
#define SZ_X2ROW  ((size_t)BT * CC * QP * 2)         // 104,857,600
#define OFF_W2B   (OFF_X2ROW + SZ_X2ROW)
#define SZ_W2B    ((size_t)CC * CC * 2)
#define OFF_WDB   (OFF_W2B + SZ_W2B)
#define SZ_WDB    ((size_t)RC * CC * 2)
#define OFF_XD    (OFF_WDB + SZ_WDB)
#define SZ_XD     ((size_t)BB * RC * TT * HWP * 4)
#define OFF_AGG   (OFF_XD + SZ_XD)

extern "C" void kernel_launch(void* const* d_in, const int* in_sizes, int n_in,
                              void* d_out, int out_size, void* d_ws, size_t ws_size,
                              hipStream_t stream) {
    (void)in_sizes; (void)n_in; (void)out_size; (void)ws_size;
    const float* x        = (const float*)d_in[0];
    const float* w_down   = (const float*)d_in[1];
    const float* w_down2  = (const float*)d_in[2];
    const float* w_sa1    = (const float*)d_in[3];
    const float* b_sa1    = (const float*)d_in[4];
    const float* w_sa2    = (const float*)d_in[5];
    const float* b_sa2    = (const float*)d_in[6];
    const float* w_sa3    = (const float*)d_in[7];
    const float* b_sa3    = (const float*)d_in[8];
    const float* w_back   = (const float*)d_in[9];
    const float* weights  = (const float*)d_in[10];
    const float* weights2 = (const float*)d_in[11];
    float* out = (float*)d_out;

    char* ws = (char*)d_ws;
    unsigned short* xbT   = (unsigned short*)(ws + OFF_XBT);
    unsigned*       xbp   = (unsigned*)(ws + OFF_XBP);
    unsigned*       x2bp  = (unsigned*)(ws + OFF_X2BP);
    unsigned short* x2row = (unsigned short*)(ws + OFF_X2ROW);
    unsigned short* w2b   = (unsigned short*)(ws + OFF_W2B);
    unsigned short* wdb   = (unsigned short*)(ws + OFF_WDB);
    float*          xd    = (float*)(ws + OFF_XD);
    float*          agg   = (float*)(ws + OFF_AGG);

    // K0: weight casts
    k_cast_w<<<(CC * CC + 255) / 256, 256, 0, stream>>>(w_down2, w_down, w2b, wdb);
    // K1: transpose/pack x (A layout + swizzled B layout)
    k_transpose<<<dim3(BT, CC / 32, (HWP + 31) / 32), dim3(32, 8), 0, stream>>>(x, xbT, xbp);
    // K2: zero x2row K-padding
    k_pad_x2row<<<(BT * CC * 8 + 255) / 256, 256, 0, stream>>>((unsigned*)x2row);
    // K3: x2 projection GEMM (WMMA, 2-wide M, pipelined)
    k_gemm_x2<<<dim3(BT, 7), 256, 0, stream>>>((const unsigned*)w2b, xbp, x2row, x2bp);
    // K4: xd projection GEMM (WMMA, pipelined)
    k_gemm_xd<<<BT, 256, 0, stream>>>((const unsigned*)wdb, xbp, xd);
    // K5: depthwise dilated convs + weighted sum
    k_conv<<<(BB * RC * TT * HWP + 255) / 256, 256, 0, stream>>>(
        xd, w_sa1, b_sa1, w_sa2, b_sa2, w_sa3, b_sa3, weights, agg);
    // K6: fused affinity + features (WMMA), writes d_out
    k_affinity<<<dim3(BT, HWP / 16), 256, 0, stream>>>(
        (const unsigned*)xbT, x2bp, (const unsigned*)x2row, weights2, out);
    // K7: gate + final elementwise multiply (in place on d_out)
    size_t ntot = (size_t)BB * CC * TT * HWP;
    k_final<<<(unsigned)((ntot + 255) / 256), 256, 0, stream>>>(out, agg, w_back);
}
